// SegCaps_42064909697732
// MI455X (gfx1250) — matrix operand order repaired and checked by reading, not compiled, not run on hardware
//
#include <hip/hip_runtime.h>
#include <math.h>

// ---------------------------------------------------------------------------
// SegCaps forward for gfx1250 (MI455X), round 3.
//  - Grouped 3D convs -> implicit GEMM, v_wmma_f32_16x16x32_f16, 4 WMMA/wave
//    per K-step (wave tile 16M x 64N), fp32 accumulate.
//  - Weights pre-packed to f16 (flip applied, zero-padded to Mp x KtotP) so
//    A-staging is contiguous -> global_load_async_to_lds_b128 (ASYNCcnt path).
//  - Activations pre-packed to f16 for the im2col gather.
//  - Routing/softmax/squash: wave32 VALU kernel, one wave per voxel.
// ---------------------------------------------------------------------------

typedef _Float16 f16;
typedef _Float16 v8h  __attribute__((ext_vector_type(8)));
typedef _Float16 v16h __attribute__((ext_vector_type(16)));
typedef float    v8f  __attribute__((ext_vector_type(8)));
typedef int      v4i  __attribute__((ext_vector_type(4)));

#define SP 4096   // 16^3 voxels, all layers

// ---- CDNA5 async global->LDS copy (guarded; falls back to vector copy) ----
#if defined(__has_builtin)
#  if __has_builtin(__builtin_amdgcn_global_load_async_to_lds_b128)
#    define HAVE_ASYNC_LDS 1
#  endif
#endif
#ifndef HAVE_ASYNC_LDS
#  define HAVE_ASYNC_LDS 0
#endif

// AS1 prints as __device__, AS3 as __shared__ in clang's HIP diagnostics;
// the builtin wants int4 pointers in those address spaces.
typedef __attribute__((address_space(1))) v4i g_v4i;
typedef __attribute__((address_space(3))) v4i l_v4i;

__device__ __forceinline__ void async_copy_b128(const f16* gsrc, f16* ldst) {
#if HAVE_ASYNC_LDS
  __builtin_amdgcn_global_load_async_to_lds_b128(
      (g_v4i*)(void*)gsrc, (l_v4i*)(void*)ldst, 0, 0);
#else
  *(v8h*)ldst = *(const v8h*)gsrc;
#endif
}

__device__ __forceinline__ void wait_async0() {
#if HAVE_ASYNC_LDS
#  if __has_builtin(__builtin_amdgcn_s_wait_asynccnt)
  __builtin_amdgcn_s_wait_asynccnt(0);
#  else
  asm volatile("s_wait_asynccnt 0x0" ::: "memory");
#  endif
#endif
}

__device__ __forceinline__ float wred32(float x) {
  #pragma unroll
  for (int off = 16; off > 0; off >>= 1) x += __shfl_xor(x, off, 32);
  return x;
}

// ---------------- stem conv: (4,14^3) -> relu -> (16,16^3), pad=2 ----------
__global__ __launch_bounds__(256)
void conv1_relu_kernel(const float* __restrict__ x,
                       const float* __restrict__ w,
                       float* __restrict__ out) {
  int idx = blockIdx.x * blockDim.x + threadIdx.x;   // 16 * 4096
  int ch  = idx >> 12;
  int vox = idx & 4095;
  int d = vox >> 8, h = (vox >> 4) & 15, wc = vox & 15;
  float acc = 0.f;
  #pragma unroll
  for (int c = 0; c < 4; ++c) {
    #pragma unroll
    for (int t = 0; t < 27; ++t) {
      int kd = t / 9, kh = (t / 3) % 3, kw = t % 3;
      int dd = d + kd - 2, hh = h + kh - 2, ww = wc + kw - 2;
      if ((unsigned)dd < 14u && (unsigned)hh < 14u && (unsigned)ww < 14u) {
        acc += x[c * 2744 + dd * 196 + hh * 14 + ww] *
               w[(ch * 4 + c) * 27 + t];
      }
    }
  }
  out[ch * SP + vox] = fmaxf(acc, 0.f);
}

// ---------------- weight pack: f32 (t0,M,z0*27) -> f16 (t0,Mp,KtotP) -------
// Deconv tap flip applied here; rows m>=M and cols k>=Ktot zero-filled so the
// conv kernel's async staging needs no guards.
__global__ __launch_bounds__(256)
void pack_weights_kernel(const float* __restrict__ w, f16* __restrict__ o,
                         int M, int Mp, int Ktot, int KtotP, int flip) {
  int idx = blockIdx.x * blockDim.x + threadIdx.x;   // t0*Mp*KtotP
  int g = idx / (Mp * KtotP);
  int r = idx - g * (Mp * KtotP);
  int m = r / KtotP;
  int k = r - m * KtotP;
  float v = 0.f;
  if (m < M && k < Ktot) {
    int c = k / 27, tap = k - c * 27;
    int tw = flip ? (26 - tap) : tap;
    v = w[((size_t)g * M + m) * Ktot + c * 27 + tw];
  }
  o[idx] = (f16)v;
}

// ---------------- activation pack: f32 -> f16 ------------------------------
__global__ __launch_bounds__(256)
void pack_act_kernel(const float* __restrict__ a, f16* __restrict__ o) {
  int idx = blockIdx.x * blockDim.x + threadIdx.x;
  o[idx] = (f16)a[idx];
}

// ---------------- grouped conv as implicit GEMM with WMMA ------------------
// actH: (t0, z0, SP) f16   wgtH: (t0, Mp, KtotP) f16   uhat: (t0, M, SP) f32
// Block = 128 thr (4 waves). Block tile 64(M) x 64(N); wave w owns rows
// [16w, 16w+16) x all 64 N. 4 WMMA accumulators per wave.
__global__ __launch_bounds__(128)
void caps_conv_wmma(const f16* __restrict__ actH,
                    const f16* __restrict__ wgtH,
                    float* __restrict__ uhat,
                    int M, int Mp, int z0, int Ktot, int KtotP) {
  __shared__ __align__(16) f16 ldsA[64 * 32];   // [m][k]
  __shared__ __align__(16) f16 ldsB[64 * 32];   // [n][k]  (transposed im2col)

  const int tid   = threadIdx.x;
  const int lane  = tid & 31;
  const int wave  = tid >> 5;
  const int nbase = blockIdx.x * 64;
  const int mblk  = blockIdx.y * 64;
  const int g     = blockIdx.z;

  const f16* actg = actH + (size_t)g * z0 * SP;

  v8f acc0 = {}, acc1 = {}, acc2 = {}, acc3 = {};

  // WMMA fragment addressing (16-bit A 16x32 / B 32x16 layouts, wave32)
  const int rowA   = wave * 16 + (lane & 15);
  const int kbaseA = (lane >= 16) ? 8 : 0;   // A: K {0..7,16..23}/{8..15,24..31}
  const int nB0    = lane & 15;
  const int kbaseB = (lane >= 16) ? 8 : 0;   // B: K {0..7,16..23}/{8..15,24..31}

  // staging coords: thread covers one row (m or n) half (16 halfs)
  const int mA  = tid >> 1;            // 0..63
  const int kH0 = (tid & 1) * 16;      // 0 / 16
  const f16* wrow = wgtH + ((size_t)g * Mp + mblk + mA) * KtotP + kH0;

  const int nS  = tid >> 1;            // 0..63
  const int vox = nbase + nS;
  const int d = vox >> 8, h = (vox >> 4) & 15, wq = vox & 15;

  const int nsteps = KtotP >> 5;
  for (int s = 0; s < nsteps; ++s) {
    const int kb = s << 5;

    // ---- stage A: async DMA of packed f16 weights into LDS (no guards) ----
    {
      const f16* src = wrow + kb;
      f16* dst = &ldsA[mA * 32 + kH0];
      async_copy_b128(src, dst);
      async_copy_b128(src + 8, dst + 8);
    }

    // ---- stage B: f16 im2col gather (incremental c/tap, zero-pad) ----
    {
      int k0 = kb + kH0;
      int c = k0 / 27;
      int tap = k0 - c * 27;
      #pragma unroll
      for (int i = 0; i < 16; ++i) {
        f16 v = (f16)0.f;
        if (k0 + i < Ktot) {
          int kd = tap / 9, kh = (tap / 3) % 3, kw = tap % 3;
          int dd = d + kd - 1, hh = h + kh - 1, ww = wq + kw - 1;
          if ((unsigned)dd < 16u && (unsigned)hh < 16u && (unsigned)ww < 16u)
            v = actg[((size_t)c << 12) + (dd << 8) + (hh << 4) + ww];
        }
        ldsB[nS * 32 + kH0 + i] = v;
        if (++tap == 27) { tap = 0; ++c; }
      }
    }

    wait_async0();
    __syncthreads();

    // ---- compute: 1 A fragment, 4 B fragments, 4 WMMAs (EXEC all ones) ----
    v8h alo = *(const v8h*)&ldsA[rowA * 32 + kbaseA];
    v8h ahi = *(const v8h*)&ldsA[rowA * 32 + kbaseA + 16];
    v16h a;
    #pragma unroll
    for (int i = 0; i < 8; ++i) { a[i] = alo[i]; a[i + 8] = ahi[i]; }

    #define B_FRAG(bn, sub)                                                  \
      v16h bn;                                                               \
      {                                                                      \
        v8h blo = *(const v8h*)&ldsB[((sub) * 16 + nB0) * 32 + kbaseB];      \
        v8h bhi = *(const v8h*)&ldsB[((sub) * 16 + nB0) * 32 + kbaseB + 16]; \
        _Pragma("unroll")                                                    \
        for (int i = 0; i < 8; ++i) { bn[i] = blo[i]; bn[i + 8] = bhi[i]; }  \
      }
    B_FRAG(b0, 0)
    acc0 = __builtin_amdgcn_wmma_f32_16x16x32_f16(false, a, false, b0,
                                                  (short)0, acc0, false, false);
    B_FRAG(b1, 1)
    acc1 = __builtin_amdgcn_wmma_f32_16x16x32_f16(false, a, false, b1,
                                                  (short)0, acc1, false, false);
    B_FRAG(b2, 2)
    acc2 = __builtin_amdgcn_wmma_f32_16x16x32_f16(false, a, false, b2,
                                                  (short)0, acc2, false, false);
    B_FRAG(b3, 3)
    acc3 = __builtin_amdgcn_wmma_f32_16x16x32_f16(false, a, false, b3,
                                                  (short)0, acc3, false, false);
    #undef B_FRAG

    __syncthreads();
  }

  // ---- writeback (C/D layout: VGPR v -> M = v + 8*(lane>=16), N = lane&15)
  const int mw   = mblk + wave * 16 + ((lane >= 16) ? 8 : 0);
  const int ncol = lane & 15;
  #pragma unroll
  for (int v = 0; v < 8; ++v) {
    int m = mw + v;
    if (m < M) {
      float* orow = uhat + ((size_t)g * M + m) * SP + nbase + ncol;
      orow[0]  = acc0[v];
      orow[16] = acc1[v];
      orow[32] = acc2[v];
      orow[48] = acc3[v];
    }
  }
}

// ---------------- dynamic routing + squash --------------------------------
// uhat: (t0, t1, z1, SP)   vout: (t1, z1, SP)
// One wave per voxel; lanes cover z1 (<=64 -> 2 per lane); b[t0][t1] in LDS.
#define MAXT1 8
__global__ __launch_bounds__(256)
void routing_kernel(const float* __restrict__ uhat,
                    float* __restrict__ vout,
                    int t0, int t1, int z1, int R) {
  __shared__ float bsh[8 * 64];                 // per-wave b[t0*t1] (<=64)
  const int tid  = threadIdx.x;
  const int lane = tid & 31;
  const int wave = tid >> 5;
  const int vox  = blockIdx.x * 8 + wave;
  float* bw = &bsh[wave * 64];

  bsh[tid] = 0.f; bsh[tid + 256] = 0.f;
  __syncthreads();

  const int zpl = (z1 + 31) >> 5;               // z per lane: 1 or 2
  for (int it = 0; it < R; ++it) {
    float sacc[MAXT1][2];
    #pragma unroll
    for (int tt = 0; tt < MAXT1; ++tt) { sacc[tt][0] = 0.f; sacc[tt][1] = 0.f; }

    for (int t = 0; t < t0; ++t) {
      float mx = -3.4e38f;
      #pragma unroll
      for (int tt = 0; tt < MAXT1; ++tt)
        if (tt < t1) mx = fmaxf(mx, bw[t * t1 + tt]);
      float cc[MAXT1]; float ssum = 0.f;
      #pragma unroll
      for (int tt = 0; tt < MAXT1; ++tt)
        if (tt < t1) { cc[tt] = __expf(bw[t * t1 + tt] - mx); ssum += cc[tt]; }
      float inv = 1.f / ssum;
      #pragma unroll
      for (int tt = 0; tt < MAXT1; ++tt) {
        if (tt >= t1) continue;
        float c = cc[tt] * inv;
        const float* up = uhat + (((size_t)t * t1 + tt) * z1) * SP + vox;
        #pragma unroll
        for (int j = 0; j < 2; ++j) {
          int z = lane + 32 * j;
          if (j < zpl && z < z1) sacc[tt][j] += c * up[(size_t)z * SP];
        }
      }
    }

    // squash over z1
    float vcap[MAXT1][2];
    #pragma unroll
    for (int tt = 0; tt < MAXT1; ++tt) {
      vcap[tt][0] = 0.f; vcap[tt][1] = 0.f;
      if (tt >= t1) continue;
      float p  = sacc[tt][0] * sacc[tt][0] + sacc[tt][1] * sacc[tt][1];
      float n2 = wred32(p);
      float sc = n2 / (1.f + n2) * rsqrtf(n2 + 1e-9f);
      vcap[tt][0] = sacc[tt][0] * sc;
      vcap[tt][1] = sacc[tt][1] * sc;
    }

    if (it < R - 1) {
      for (int t = 0; t < t0; ++t) {
        #pragma unroll
        for (int tt = 0; tt < MAXT1; ++tt) {
          if (tt >= t1) continue;
          const float* up = uhat + (((size_t)t * t1 + tt) * z1) * SP + vox;
          float p = 0.f;
          #pragma unroll
          for (int j = 0; j < 2; ++j) {
            int z = lane + 32 * j;
            if (j < zpl && z < z1) p += up[(size_t)z * SP] * vcap[tt][j];
          }
          float dot = wred32(p);
          if (lane == 0) bw[t * t1 + tt] += dot;
        }
      }
    } else {
      #pragma unroll
      for (int tt = 0; tt < MAXT1; ++tt) {
        if (tt >= t1) continue;
        #pragma unroll
        for (int j = 0; j < 2; ++j) {
          int z = lane + 32 * j;
          if (j < zpl && z < z1)
            vout[((size_t)tt * z1 + z) * SP + vox] = vcap[tt][j];
        }
      }
    }
    __syncthreads();
  }
}

// ---------------- final L2 norm over 16 channels --------------------------
__global__ __launch_bounds__(256)
void final_norm_kernel(const float* __restrict__ v, float* __restrict__ out) {
  int idx = blockIdx.x * blockDim.x + threadIdx.x;
  if (idx < SP) {
    float s = 0.f;
    #pragma unroll
    for (int z = 0; z < 16; ++z) { float a = v[z * SP + idx]; s += a * a; }
    out[idx] = sqrtf(s + 1e-9f);
  }
}

// ---------------------------------------------------------------------------
extern "C" void kernel_launch(void* const* d_in, const int* in_sizes, int n_in,
                              void* d_out, int out_size, void* d_ws, size_t ws_size,
                              hipStream_t stream) {
  (void)in_sizes; (void)n_in; (void)out_size; (void)ws_size;
  const float* x  = (const float*)d_in[0];
  const float* wc = (const float*)d_in[1];
  const float* W[12];
  for (int i = 0; i < 12; ++i) W[i] = (const float*)d_in[2 + i];
  float* outf = (float*)d_out;

  float* ws  = (float*)d_ws;
  const size_t sp = SP;
  float* C10 = ws;                     // (3,16,sp): [L8 out(2) | skip1]
  float* V1a = C10 + 3 * 16 * sp;      // (2,16,sp)
  float* C7  = V1a + 2 * 16 * sp;      // (8,16,sp): [L6 out(4) | skip2(4)]
  float* V2a = C7  + 8 * 16 * sp;      // (4,32,sp)
  float* C5  = V2a + 4 * 32 * sp;      // (16,32,sp): [L4 out(8) | skip3(8)]
  float* V3a = C5  + 16 * 32 * sp;     // (8,64,sp)
  float* V3b = V3a + 8 * 64 * sp;      // (8,32,sp)
  float* V5  = V3b + 8 * 32 * sp;      // (4,32,sp)
  float* V7  = V5  + 4 * 32 * sp;      // (4,16,sp)
  float* V10 = V7  + 4 * 16 * sp;      // (1,16,sp)
  float* UH  = V10 + 1 * 16 * sp;      // u_hat scratch: 8*512*sp floats max
  f16*   WPH = (f16*)(UH + (size_t)8 * 512 * sp);  // packed weights (<=3.6M halfs)
  f16*   AH  = WPH + (size_t)4 * 1024 * 1024;      // packed act (<=2.1M halfs)

  float* u0    = C10 + 2 * 16 * sp;    // skip1 slot (also layer1a input)
  float* skip2 = C7  + 4 * 16 * sp;
  float* skip3 = C5  + 8 * 32 * sp;

  conv1_relu_kernel<<<dim3(16 * SP / 256), 256, 0, stream>>>(x, wc, u0);

  auto caps = [&](const float* in, const float* w, float* vout,
                  int t0, int z0, int t1, int z1, int R, int flip) {
    int M = t1 * z1, Ktot = z0 * 27;
    int Mp = ((M + 63) / 64) * 64;
    int KtotP = ((Ktot + 31) / 32) * 32;
    pack_weights_kernel<<<dim3(t0 * Mp * KtotP / 256), 256, 0, stream>>>(
        w, WPH, M, Mp, Ktot, KtotP, flip);
    pack_act_kernel<<<dim3(t0 * z0 * SP / 256), 256, 0, stream>>>(in, AH);
    dim3 grid(SP / 64, Mp / 64, t0);
    caps_conv_wmma<<<grid, 128, 0, stream>>>(AH, WPH, UH, M, Mp, z0, Ktot, KtotP);
    routing_kernel<<<dim3(SP / 8), 256, 0, stream>>>(UH, vout, t0, t1, z1, R);
  };

  caps(u0,    W[0],  V1a,    1, 16, 2, 16, 1, 0);   // w1a
  caps(V1a,   W[1],  skip2,  2, 16, 4, 16, 3, 0);   // w1b  (-> skip2 slot)
  caps(skip2, W[2],  V2a,    4, 16, 4, 32, 3, 0);   // w2a
  caps(V2a,   W[3],  skip3,  4, 32, 8, 32, 3, 0);   // w2b  (-> skip3 slot)
  caps(skip3, W[4],  V3a,    8, 32, 8, 64, 3, 0);   // w3a
  caps(V3a,   W[5],  V3b,    8, 64, 8, 32, 3, 0);   // w3b
  caps(V3b,   W[6],  C5,     8, 32, 8, 32, 3, 1);   // w4  deconv -> concat skip3
  caps(C5,    W[7],  V5,    16, 32, 4, 32, 3, 0);   // w5
  caps(V5,    W[8],  C7,     4, 32, 4, 16, 3, 1);   // w6  deconv -> concat skip2
  caps(C7,    W[9],  V7,     8, 16, 4, 16, 3, 0);   // w7
  caps(V7,    W[10], C10,    4, 16, 2, 16, 3, 1);   // w8  deconv -> concat skip1
  caps(C10,   W[11], V10,    3, 16, 1, 16, 3, 0);   // w10

  final_norm_kernel<<<dim3((SP + 255) / 256), 256, 0, stream>>>(V10, outf);
}
